// NTXentLoss_76733885710860
// MI455X (gfx1250) — matrix operand order repaired and checked by reading, not compiled, not run on hardware
//
#include <hip/hip_runtime.h>
#include <hip/hip_bf16.h>

typedef __attribute__((ext_vector_type(16))) _Float16 v16h;
typedef __attribute__((ext_vector_type(8)))  _Float16 v8h;
typedef __attribute__((ext_vector_type(8)))  float    v8f;
typedef __attribute__((ext_vector_type(4)))  float    v4f;

#define NTOT   8192
#define BHALF  4096
#define DDIM   256
// Both WMMA operands are pre-scaled by sqrt(2/ln2), so the accumulator is
// directly sim * (2/ln2) == the exp2 argument for exp(sim/tau), tau = 0.5.
#define PRESCALE 1.69864368f   // sqrt(2.885390081777927)

// ---------------------------------------------------------------------------
// Kernel 1: row norms + pre-scaled f16 normalized matrix; zero rowsum[], acc.
// One wave per row; each lane handles 8 contiguous floats.
// ---------------------------------------------------------------------------
__global__ __launch_bounds__(256) void ntx_norm_kernel(
    const float* __restrict__ zi, const float* __restrict__ zj,
    _Float16* __restrict__ znh, float* __restrict__ norms,
    float* __restrict__ rowsum, float* __restrict__ acc)
{
    const int wave = threadIdx.x >> 5;
    const int lane = threadIdx.x & 31;
    const int row  = blockIdx.x * 8 + wave;
    const float* src = (row < BHALF) ? (zi + row * DDIM)
                                     : (zj + (row - BHALF) * DDIM);
    const v4f* p = (const v4f*)(src + lane * 8);
    v4f x0 = p[0], x1 = p[1];
    float ss = 0.f;
#pragma unroll
    for (int i = 0; i < 4; ++i) ss += x0[i] * x0[i] + x1[i] * x1[i];
#pragma unroll
    for (int off = 16; off; off >>= 1) ss += __shfl_xor(ss, off, 32);
    float nrm = fmaxf(sqrtf(ss), 1e-8f);
    float inv = PRESCALE / nrm;
    v8h o;
#pragma unroll
    for (int i = 0; i < 4; ++i) {
        o[i]     = (_Float16)(x0[i] * inv);
        o[i + 4] = (_Float16)(x1[i] * inv);
    }
    *(v8h*)(znh + row * DDIM + lane * 8) = o;
    if (lane == 0) { norms[row] = nrm; rowsum[row] = 0.f; }
    if (blockIdx.x == 0 && threadIdx.x == 0) acc[0] = 0.f;
}

// ---------------------------------------------------------------------------
// Kernel 2: positives in f32: pos[r] = dot(zn[r], zn[r^4096]) / tau.
// ---------------------------------------------------------------------------
__global__ __launch_bounds__(256) void ntx_pos_kernel(
    const float* __restrict__ zi, const float* __restrict__ zj,
    const float* __restrict__ norms, float* __restrict__ pos)
{
    const int wave = threadIdx.x >> 5;
    const int lane = threadIdx.x & 31;
    const int row  = blockIdx.x * 8 + wave;
    const int par  = row ^ BHALF;
    const float* a = (row < BHALF) ? (zi + row * DDIM) : (zj + (row - BHALF) * DDIM);
    const float* b = (par < BHALF) ? (zi + par * DDIM) : (zj + (par - BHALF) * DDIM);
    const v4f* pa = (const v4f*)(a + lane * 8);
    const v4f* pb = (const v4f*)(b + lane * 8);
    v4f a0 = pa[0], a1 = pa[1], b0 = pb[0], b1 = pb[1];
    float dot = 0.f;
#pragma unroll
    for (int i = 0; i < 4; ++i) dot += a0[i] * b0[i] + a1[i] * b1[i];
#pragma unroll
    for (int off = 16; off; off >>= 1) dot += __shfl_xor(dot, off, 32);
    if (lane == 0) pos[row] = dot / (norms[row] * norms[par]) * 2.0f;  // /tau
}

// ---------------------------------------------------------------------------
// Kernel 3: fused GEMM + masked sum-of-exp via WMMA f16 16x16x32.
// grid = (8 column chunks, 64 row blocks); 8 waves/block, 16 rows per wave.
// A tile (16 rows x 256) held in registers (8 x v16h) for the whole sweep.
// Logits bounded (|c| <= 2.89) -> raw v_exp_f32, no max tracking; partial
// sums are additive across blocks -> combined via global f32 atomics.
// ---------------------------------------------------------------------------
__global__ __launch_bounds__(256) void ntx_lse_kernel(
    const _Float16* __restrict__ znh, float* __restrict__ rowsum)
{
    const int lane  = threadIdx.x & 31;
    const int wave  = threadIdx.x >> 5;
    const int half  = lane >> 4;      // 0: lanes 0-15, 1: lanes 16-31
    const int nlane = lane & 15;      // column index within 16-wide tile
    const int rowBase = blockIdx.y * 128 + wave * 16;

    // ---- A fragments: row M = rowBase + nlane, per ISA 16-bit A 16x32 layout
    const _Float16* abase = znh + (rowBase + nlane) * DDIM + half * 8;
    v16h a[8];
#pragma unroll
    for (int kb = 0; kb < 8; ++kb) {
        v8h lo = *(const v8h*)(abase + kb * 32);        // K = kb*32 + half*8 + 0..7
        v8h hi = *(const v8h*)(abase + kb * 32 + 16);   // K = kb*32 + 16 + half*8 + 0..7
        a[kb] = __builtin_shufflevector(lo, hi, 0, 1, 2, 3, 4, 5, 6, 7,
                                                8, 9, 10, 11, 12, 13, 14, 15);
    }

    // Diagonal mask: C/D layout VGPR vg, lanes 0-15 -> M = vg, lanes 16-31 -> M = 8+vg
    const int m0 = half * 8;
    const int diagVg = nlane - m0;
    const unsigned diagBit = (diagVg >= 0 && diagVg < 8) ? (1u << diagVg) : 0u;

    float sums[8];
#pragma unroll
    for (int vg = 0; vg < 8; ++vg) sums[vg] = 0.f;

    const int tBeg = blockIdx.x * 64;
    const int tEnd = tBeg + 64;
    for (int t = tBeg; t < tEnd; ++t) {
        const int colBase = t * 16;
        // B fragment: column N = zn row (colBase + nlane); K split like A (half*8, 16+half*8)
        const _Float16* bp = znh + (colBase + nlane) * DDIM + half * 8;
        v8f c = {};
#pragma unroll
        for (int kb = 0; kb < 8; ++kb) {
            v8h lo = *(const v8h*)(bp + kb * 32);       // K = kb*32 + half*8 + 0..7
            v8h hi = *(const v8h*)(bp + kb * 32 + 16);  // K = kb*32 + 16 + half*8 + 0..7
            v16h b = __builtin_shufflevector(lo, hi, 0, 1, 2, 3, 4, 5, 6, 7,
                                                     8, 9, 10, 11, 12, 13, 14, 15);
            c = __builtin_amdgcn_wmma_f32_16x16x32_f16(
                    false, a[kb], false, b, (short)0, c, false, false);
        }
        const unsigned dm = (colBase == rowBase) ? diagBit : 0u;
#pragma unroll
        for (int vg = 0; vg < 8; ++vg) {
            // c[vg] is already sim * 2/ln2; args bounded -> raw v_exp_f32
            float e = __builtin_amdgcn_exp2f(c[vg]);
            if (dm & (1u << vg)) e = 0.f;               // mask self-similarity
            sums[vg] += e;
        }
    }

    // Reduce across the 16 lanes of each half (same row, different columns)
#pragma unroll
    for (int vg = 0; vg < 8; ++vg) {
        float s = sums[vg];
#pragma unroll
        for (int off = 8; off; off >>= 1) s += __shfl_xor(s, off, 16);
        sums[vg] = s;
    }
    if (nlane == 0) {
#pragma unroll
        for (int vg = 0; vg < 8; ++vg)
            atomicAdd(&rowsum[rowBase + m0 + vg], sums[vg]);
    }
}

// ---------------------------------------------------------------------------
// Kernel 4: per-row loss contribution, wave-reduced, atomically accumulated.
// ---------------------------------------------------------------------------
__global__ __launch_bounds__(256) void ntx_loss_kernel(
    const float* __restrict__ rowsum, const float* __restrict__ pos,
    float* __restrict__ acc)
{
    const int r = blockIdx.x * 256 + threadIdx.x;
    float contrib = __logf(rowsum[r]) - pos[r];
#pragma unroll
    for (int off = 16; off; off >>= 1) contrib += __shfl_xor(contrib, off, 32);
    if ((threadIdx.x & 31) == 0) atomicAdd(acc, contrib);
}

__global__ void ntx_fin_kernel(const float* __restrict__ acc, float* __restrict__ out)
{
    out[0] = acc[0] * (1.0f / (float)NTOT);
}

// ---------------------------------------------------------------------------
extern "C" void kernel_launch(void* const* d_in, const int* in_sizes, int n_in,
                              void* d_out, int out_size, void* d_ws, size_t ws_size,
                              hipStream_t stream)
{
    const float* zi = (const float*)d_in[0];
    const float* zj = (const float*)d_in[1];

    char* ws = (char*)d_ws;
    _Float16* znh    = (_Float16*)ws;                                 // 4 MiB
    float*    norms  = (float*)(ws + (size_t)NTOT * DDIM * sizeof(_Float16));
    float*    pos    = norms + NTOT;
    float*    rowsum = pos + NTOT;
    float*    acc    = rowsum + NTOT;

    ntx_norm_kernel<<<NTOT / 8, 256, 0, stream>>>(zi, zj, znh, norms, rowsum, acc);
    ntx_pos_kernel <<<NTOT / 8, 256, 0, stream>>>(zi, zj, norms, pos);
    ntx_lse_kernel <<<dim3(8, 64), 256, 0, stream>>>(znh, rowsum);
    ntx_loss_kernel<<<NTOT / 256, 256, 0, stream>>>(rowsum, pos, acc);
    ntx_fin_kernel <<<1, 1, 0, stream>>>(acc, (float*)d_out);
}
